// GroupedQueryAttention_41016937677093
// MI455X (gfx1250) — matrix operand order repaired and checked by reading, not compiled, not run on hardware
//
#include <hip/hip_runtime.h>
#include <math.h>

// ---------------------------------------------------------------------------
// GQA for MI455X (gfx1250): compute-bound -> bf16 WMMA (v_wmma_f32_16x16x32_bf16)
// 128x128x32 tiles, 256 threads (8 wave32), double-buffered LDS, wave grid 4x2,
// each wave owns a 32x64 sub-tile = 2x4 WMMA accumulators (16x16, f32).
// Staging uses GLOBAL_LOAD_ASYNC_TO_LDS_B128 (ASYNCcnt) when the toolchain
// exposes the builtins; falls back to global_load_b128 + ds_store_b128.
// ---------------------------------------------------------------------------

typedef __bf16 bf16_t;
typedef __attribute__((ext_vector_type(8)))  __bf16 v8bf;
typedef __attribute__((ext_vector_type(16))) __bf16 v16bf;
typedef __attribute__((ext_vector_type(8)))  float  v8f;
typedef __attribute__((ext_vector_type(4)))  int    v4i;

#define BM 128
#define BN 128
#define BK 32
#define BKP 40   // padded LDS row length in elements (80B rows, 16B aligned chunks)

#if __has_builtin(__builtin_amdgcn_global_load_async_to_lds_b128) && \
    __has_builtin(__builtin_amdgcn_s_wait_asynccnt)
#define HAS_ASYNC_LDS 1
#else
#define HAS_ASYNC_LDS 0
#endif

#if HAS_ASYNC_LDS
#define AS_GLOBAL __attribute__((address_space(1)))
#define AS_LOCAL  __attribute__((address_space(3)))
// one wave32 lane copies 16B global -> LDS via the async DMA path
__device__ __forceinline__ void async_copy16(const unsigned short* g,
                                             unsigned short* l) {
  __builtin_amdgcn_global_load_async_to_lds_b128(
      (AS_GLOBAL v4i*)g, (AS_LOCAL v4i*)l, 0, 0);
}
__device__ __forceinline__ void async_wait0() {
  __builtin_amdgcn_s_wait_asynccnt(0);
}
#else
__device__ __forceinline__ void async_wait0() {}
#endif

__device__ __forceinline__ unsigned short f32_to_bf16(float f) {
  unsigned int u = __float_as_uint(f);
  u += 0x7FFFu + ((u >> 16) & 1u);          // round-to-nearest-even
  return (unsigned short)(u >> 16);
}

// Combine two 16B LDS chunks into one 32B WMMA fragment (two ds_load_b128).
__device__ __forceinline__ v16bf frag2(const unsigned short* p0,
                                       const unsigned short* p1) {
  v8bf lo = *(const v8bf*)p0;
  v8bf hi = *(const v8bf*)p1;
  return __builtin_shufflevector(lo, hi, 0, 1, 2, 3, 4, 5, 6, 7,
                                         8, 9, 10, 11, 12, 13, 14, 15);
}

// ---------------------------------------------------------------------------
// cast f32 -> bf16 (grid-stride)
// ---------------------------------------------------------------------------
__global__ __launch_bounds__(256) void cast_f32_bf16(
    const float* __restrict__ in, unsigned short* __restrict__ out, long n) {
  long i = (long)blockIdx.x * blockDim.x + threadIdx.x;
  long stride = (long)gridDim.x * blockDim.x;
  for (; i < n; i += stride) out[i] = f32_to_bf16(in[i]);
}

// ---------------------------------------------------------------------------
// C[M,N] = A[M,K] * B[N,K]^T (+bias). A,B bf16 row-major. OUT bf16 or f32.
// REMAP_B: B's k-index d -> ((d/128)%4)*128 + d%128  (GQA group tiling)
// CAUSAL : skip tiles entirely above the diagonal (scores GEMM)
// ---------------------------------------------------------------------------
template <bool OUT_BF16, bool BIAS, bool REMAP_B, bool CAUSAL>
__global__ __launch_bounds__(256) void gemm_nt(
    const unsigned short* __restrict__ A, const unsigned short* __restrict__ Bm,
    const float* __restrict__ bias, void* __restrict__ Cv,
    int K, int ldb, int ldc) {
  __shared__ unsigned short lA[2][BM][BKP];
  __shared__ unsigned short lB[2][BN][BKP];

  const int tid = threadIdx.x;
  const int lane = tid & 31;
  const int wid = tid >> 5;
  const int lrow = lane & 15;   // M (A frag) / N (B frag, C cols)
  const int khalf = lane >> 4;  // k-half selector; C row offset *8
  const int waveM = wid & 3;    // 4 waves x 32 rows
  const int waveN = wid >> 2;   // 2 waves x 64 cols

  const int n0 = blockIdx.x * BN;
  const int m0 = blockIdx.y * BM;
  if (CAUSAL && n0 > m0 + BM - 1) return;  // fully masked tile

  const int lda = K;
  const int nk = K / BK;

  v8f acc[2][4] = {};

  // Issue the tile copy for LDS buffer `buf`, k-window kk0.
  auto stage_issue = [&](int buf, int kk0) {
#pragma unroll
    for (int i = 0; i < 2; ++i) {
      int c = tid + i * 256;      // 0..511 chunks of 8 bf16
      int row = c >> 2;           // 0..127
      int c8 = (c & 3) << 3;      // 0,8,16,24
      const unsigned short* ga = A + (size_t)(m0 + row) * lda + (kk0 + c8);
      int kcol = kk0 + c8;
      if (REMAP_B) { int head = kcol >> 7; kcol = ((head & 3) << 7) | (kcol & 127); }
      const unsigned short* gb = Bm + (size_t)(n0 + row) * ldb + kcol;
#if HAS_ASYNC_LDS
      async_copy16(ga, &lA[buf][row][c8]);
      async_copy16(gb, &lB[buf][row][c8]);
#else
      uint4 va = *(const uint4*)ga;
      *(uint4*)&lA[buf][row][c8] = va;
      uint4 vb = *(const uint4*)gb;
      *(uint4*)&lB[buf][row][c8] = vb;
#endif
    }
  };

  stage_issue(0, 0);
  async_wait0();
  __syncthreads();

  for (int kt = 0; kt < nk; ++kt) {
    const int cur = kt & 1;
    // issue DMA for next tile first so it overlaps the WMMA stream
    if (kt + 1 < nk) stage_issue(cur ^ 1, (kt + 1) * BK);
    if (kt + 2 < nk)  // global_prefetch_b8 for the tile after next
      __builtin_prefetch(A + (size_t)(m0 + (tid >> 2)) * lda + (kt + 2) * BK, 0, 0);

    v16bf af[2], bfr[4];
#pragma unroll
    for (int tm = 0; tm < 2; ++tm) {
      const unsigned short* p = &lA[cur][waveM * 32 + tm * 16 + lrow][khalf * 8];
      af[tm] = frag2(p, p + 16);  // ISA A layout: k in {kh*8..+8} U {16+kh*8..+8}
    }
#pragma unroll
    for (int tn = 0; tn < 4; ++tn) {
      const unsigned short* p = &lB[cur][waveN * 64 + tn * 16 + lrow][khalf * 8];
      bfr[tn] = frag2(p, p + 16);  // ISA B layout: k in {kh*8..+8} U {16+kh*8..+8}
    }
#pragma unroll
    for (int tm = 0; tm < 2; ++tm)
#pragma unroll
      for (int tn = 0; tn < 4; ++tn)
        acc[tm][tn] = __builtin_amdgcn_wmma_f32_16x16x32_bf16(
            false, af[tm], false, bfr[tn], (short)0, acc[tm][tn], false, false);

    if (kt + 1 < nk) async_wait0();
    __syncthreads();
  }

#pragma unroll
  for (int tm = 0; tm < 2; ++tm)
#pragma unroll
    for (int tn = 0; tn < 4; ++tn) {
      const int gm0 = m0 + waveM * 32 + tm * 16 + khalf * 8;
      const int gn = n0 + waveN * 64 + tn * 16 + lrow;
      float badd = BIAS ? bias[gn] : 0.0f;
#pragma unroll
      for (int r = 0; r < 8; ++r) {
        float v = acc[tm][tn][r] + badd;
        size_t idx = (size_t)(gm0 + r) * ldc + gn;
        if (OUT_BF16) ((unsigned short*)Cv)[idx] = f32_to_bf16(v);
        else          ((float*)Cv)[idx] = v;
      }
    }
}

// ---------------------------------------------------------------------------
// C[M,N] = A[M,K] * B[K,N]. B transposed into LDS so the fragment path is
// identical to gemm_nt. REMAP_BN: n -> ((n/128)%4)*128 + n%128 (V tiling).
// CAUSAL_K: truncate k-loop at the diagonal block (probs are lower-triangular).
// A tile staged async; B tile needs an in-flight transpose -> register path.
// ---------------------------------------------------------------------------
template <bool OUT_BF16, bool REMAP_BN_, bool CAUSAL_K>
__global__ __launch_bounds__(256) void gemm_nn(
    const unsigned short* __restrict__ A, const unsigned short* __restrict__ Bm,
    void* __restrict__ Cv, int K, int ldb, int ldc) {
  __shared__ unsigned short lA[2][BM][BKP];
  __shared__ unsigned short lB[2][BN][BKP];

  const int tid = threadIdx.x;
  const int lane = tid & 31;
  const int wid = tid >> 5;
  const int lrow = lane & 15;
  const int khalf = lane >> 4;
  const int waveM = wid & 3;
  const int waveN = wid >> 2;

  const int n0 = blockIdx.x * BN;
  const int m0 = blockIdx.y * BM;
  const int lda = K;
  int kEnd = K;
  if (CAUSAL_K) kEnd = (m0 + BM < K) ? (m0 + BM) : K;
  const int nk = kEnd / BK;

  v8f acc[2][4] = {};

  auto stage_issue = [&](int buf, int kk0) {
#pragma unroll
    for (int i = 0; i < 2; ++i) {
      int c = tid + i * 256;
      // A tile (same as NT): async DMA when available
      int row = c >> 2;
      int c8 = (c & 3) << 3;
      const unsigned short* ga = A + (size_t)(m0 + row) * lda + (kk0 + c8);
#if HAS_ASYNC_LDS
      async_copy16(ga, &lA[buf][row][c8]);
#else
      uint4 va = *(const uint4*)ga;
      *(uint4*)&lA[buf][row][c8] = va;
#endif
      // B tile: rows k (32) x cols n (128); transpose-store as lB[n][k]
      int kr = c >> 4;            // 0..31
      int cn = (c & 15) << 3;     // 0..120
      int gn = n0 + cn;
      if (REMAP_BN_) { int head = gn >> 7; gn = ((head & 3) << 7) | (gn & 127); }
      const unsigned short* gb = Bm + (size_t)(kk0 + kr) * ldb + gn;
      uint4 vb = *(const uint4*)gb;
      union { uint4 u; unsigned short s[8]; } t;
      t.u = vb;
#pragma unroll
      for (int j = 0; j < 8; ++j) lB[buf][cn + j][kr] = t.s[j];
    }
  };

  stage_issue(0, 0);
  async_wait0();
  __syncthreads();

  for (int kt = 0; kt < nk; ++kt) {
    const int cur = kt & 1;
    if (kt + 1 < nk) stage_issue(cur ^ 1, (kt + 1) * BK);
    if (kt + 2 < nk)
      __builtin_prefetch(A + (size_t)(m0 + (tid >> 2)) * lda + (kt + 2) * BK, 0, 0);

    v16bf af[2], bfr[4];
#pragma unroll
    for (int tm = 0; tm < 2; ++tm) {
      const unsigned short* p = &lA[cur][waveM * 32 + tm * 16 + lrow][khalf * 8];
      af[tm] = frag2(p, p + 16);
    }
#pragma unroll
    for (int tn = 0; tn < 4; ++tn) {
      const unsigned short* p = &lB[cur][waveN * 64 + tn * 16 + lrow][khalf * 8];
      bfr[tn] = frag2(p, p + 16);
    }
#pragma unroll
    for (int tm = 0; tm < 2; ++tm)
#pragma unroll
      for (int tn = 0; tn < 4; ++tn)
        acc[tm][tn] = __builtin_amdgcn_wmma_f32_16x16x32_bf16(
            false, af[tm], false, bfr[tn], (short)0, acc[tm][tn], false, false);

    if (kt + 1 < nk) async_wait0();
    __syncthreads();
  }

#pragma unroll
  for (int tm = 0; tm < 2; ++tm)
#pragma unroll
    for (int tn = 0; tn < 4; ++tn) {
      const int gm0 = m0 + waveM * 32 + tm * 16 + khalf * 8;
      const int gn = n0 + waveN * 64 + tn * 16 + lrow;
#pragma unroll
      for (int r = 0; r < 8; ++r) {
        float v = acc[tm][tn][r];
        size_t idx = (size_t)(gm0 + r) * ldc + gn;
        if (OUT_BF16) ((unsigned short*)Cv)[idx] = f32_to_bf16(v);
        else          ((float*)Cv)[idx] = v;
      }
    }
}

// ---------------------------------------------------------------------------
// Causal row softmax with post-scale (ref divides by sqrt(HS) AFTER softmax).
// One 256-thread block per row; masked tail written as zero (read by AV GEMM).
// ---------------------------------------------------------------------------
__global__ __launch_bounds__(256) void softmax_causal(
    const float* __restrict__ S, unsigned short* __restrict__ P, int T,
    float postScale) {
  const int r = blockIdx.x;
  const int tid = threadIdx.x;
  const float* row = S + (size_t)r * T;
  __shared__ float red[256];

  float mx = -INFINITY;
  for (int j = tid; j <= r; j += 256) mx = fmaxf(mx, row[j]);
  red[tid] = mx;
  __syncthreads();
  for (int s = 128; s > 0; s >>= 1) {
    if (tid < s) red[tid] = fmaxf(red[tid], red[tid + s]);
    __syncthreads();
  }
  const float rowmax = red[0];
  __syncthreads();

  float sum = 0.0f;
  for (int j = tid; j <= r; j += 256) sum += __expf(row[j] - rowmax);
  red[tid] = sum;
  __syncthreads();
  for (int s = 128; s > 0; s >>= 1) {
    if (tid < s) red[tid] += red[tid + s];
    __syncthreads();
  }
  const float inv = postScale / red[0];

  unsigned short* prow = P + (size_t)r * T;
  for (int j = tid; j < T; j += 256) {
    float v = (j <= r) ? __expf(row[j] - rowmax) * inv : 0.0f;
    prow[j] = f32_to_bf16(v);
  }
}

// ---------------------------------------------------------------------------
extern "C" void kernel_launch(void* const* d_in, const int* in_sizes, int n_in,
                              void* d_out, int out_size, void* d_ws,
                              size_t ws_size, hipStream_t stream) {
  (void)in_sizes; (void)n_in; (void)out_size; (void)ws_size;
  constexpr int Bb = 4, T = 2048, D = 2048;
  constexpr int NQ = 2048;   // NH*HS
  constexpr int NKV = 512;   // NG*HS
  constexpr int MT = Bb * T; // 8192

  const float* x  = (const float*)d_in[0];
  const float* WQ = (const float*)d_in[1];
  const float* bQ = (const float*)d_in[2];
  const float* WK = (const float*)d_in[3];
  const float* bK = (const float*)d_in[4];
  const float* WV = (const float*)d_in[5];
  const float* bV = (const float*)d_in[6];
  const float* Wp = (const float*)d_in[7];
  float* out = (float*)d_out;

  char* ws = (char*)d_ws;
  size_t off = 0;
  auto alloc = [&](size_t bytes) {
    size_t o = off;
    off += (bytes + 255) & ~(size_t)255;
    return o;
  };
  unsigned short* xh  = (unsigned short*)(ws + alloc((size_t)MT * D * 2));
  unsigned short* wqh = (unsigned short*)(ws + alloc((size_t)NQ * D * 2));
  unsigned short* wkh = (unsigned short*)(ws + alloc((size_t)NKV * D * 2));
  unsigned short* wvh = (unsigned short*)(ws + alloc((size_t)NKV * D * 2));
  unsigned short* wph = (unsigned short*)(ws + alloc((size_t)NQ * D * 2));
  unsigned short* Qh  = (unsigned short*)(ws + alloc((size_t)MT * NQ * 2));
  unsigned short* Kh  = (unsigned short*)(ws + alloc((size_t)MT * NKV * 2));
  unsigned short* Vh  = (unsigned short*)(ws + alloc((size_t)MT * NKV * 2));
  float*          Sb  = (float*)(ws + alloc((size_t)T * T * 4));   // per-batch reuse
  unsigned short* Pb  = (unsigned short*)(ws + alloc((size_t)T * T * 2));
  unsigned short* Zh  = (unsigned short*)(ws + alloc((size_t)MT * NQ * 2));

  cast_f32_bf16<<<2048, 256, 0, stream>>>(x,  xh,  (long)MT * D);
  cast_f32_bf16<<<1024, 256, 0, stream>>>(WQ, wqh, (long)NQ * D);
  cast_f32_bf16<<<512,  256, 0, stream>>>(WK, wkh, (long)NKV * D);
  cast_f32_bf16<<<512,  256, 0, stream>>>(WV, wvh, (long)NKV * D);
  cast_f32_bf16<<<1024, 256, 0, stream>>>(Wp, wph, (long)NQ * D);

  // Q/K/V projections: C = x * W^T + b  (bf16 out)
  gemm_nt<true, true, false, false>
      <<<dim3(NQ / BN, MT / BM), 256, 0, stream>>>(xh, wqh, bQ, Qh, D, D, NQ);
  gemm_nt<true, true, false, false>
      <<<dim3(NKV / BN, MT / BM), 256, 0, stream>>>(xh, wkh, bK, Kh, D, D, NKV);
  gemm_nt<true, true, false, false>
      <<<dim3(NKV / BN, MT / BM), 256, 0, stream>>>(xh, wvh, bV, Vh, D, D, NKV);

  const float postScale = 0.08838834764831845f;  // 1/sqrt(128)
  for (int b = 0; b < Bb; ++b) {
    const unsigned short* Qb = Qh + (size_t)b * T * NQ;
    const unsigned short* Kb = Kh + (size_t)b * T * NKV;
    const unsigned short* Vb = Vh + (size_t)b * T * NKV;
    unsigned short* Zb = Zh + (size_t)b * T * NQ;
    // scores[t,s] = sum_d Q[t,d] * K[s, map(d)]  (f32 out, causal tile skip)
    gemm_nt<false, false, true, true>
        <<<dim3(T / BN, T / BM), 256, 0, stream>>>(Qb, Kb, nullptr, Sb, NQ, NKV, T);
    softmax_causal<<<dim3(T), 256, 0, stream>>>(Sb, Pb, T, postScale);
    // Z[t,d] = sum_s P[t,s] * V[s, map(d)]  (bf16 out, k-loop truncated)
    gemm_nn<true, true, true>
        <<<dim3(NQ / BN, T / BM), 256, 0, stream>>>(Pb, Vb, Zb, T, NKV, NQ);
  }

  // out = Z * proj  (f32 out)
  gemm_nn<false, false, false>
      <<<dim3(D / BN, MT / BM), 256, 0, stream>>>(Zh, wph, out, NQ, D, D);
}
